// Codebook_21148418966051
// MI455X (gfx1250) — matrix operand order repaired and checked by reading, not compiled, not run on hardware
//
#include <hip/hip_runtime.h>
#include <hip/hip_bf16.h>

// ---------------------------------------------------------------------------
// VQ codebook quantization on MI455X (gfx1250).
// GEMM-argmin fused via bf16x2 split-precision emulated-f32 WMMA:
//   sim ~= Ahi*Bhi + Ahi*Blo + Alo*Bhi   (f32 accumulate, ~2^-16 rel error)
// Shapes: N=65536 rows, D=64, K=4096 codes.
// ---------------------------------------------------------------------------

typedef __attribute__((ext_vector_type(16))) __bf16 v16bf;
typedef __attribute__((ext_vector_type(8)))  float  v8f;

#define DIMD     64
#define KCODES   4096
#define NROWS    65536
#define MTOTAL   4194304      // NROWS * DIMD
#define NWAVES   2048         // NROWS / 32 rows-per-wave

// split f32 -> bf16 high part (truncate) + bf16 residual
__device__ __forceinline__ void splitf(float f, __bf16 &hi, __bf16 &lo) {
    unsigned u = __float_as_uint(f) & 0xFFFF0000u;
    float fh = __uint_as_float(u);
    hi = (__bf16)fh;            // exact: fh already bf16-representable
    lo = (__bf16)(f - fh);      // residual, RNE
}

// ---------------------------------------------------------------------------
// Kernel 0: repack codebook (D=64 x K=4096, row-major f32) into split-bf16
// WMMA B-operand layout. For K-tile t (16 codes) and D-chunk c (32 dims):
//   lane = ((d>>3)&1)*16 + (k&15),  element i = (d&7) + ((d&16)>>1)
// stored contiguously so the hot loop loads one v16bf (32B) per lane.
// ---------------------------------------------------------------------------
__global__ __launch_bounds__(256) void pack_embedding(
    const float* __restrict__ emb, __bf16* __restrict__ bhi, __bf16* __restrict__ blo)
{
    int idx = blockIdx.x * blockDim.x + threadIdx.x;     // 0 .. 64*4096-1
    if (idx >= DIMD * KCODES) return;
    int d = idx >> 12;          // emb is (64, 4096) row-major
    int k = idx & 4095;
    __bf16 h, l;
    splitf(emb[idx], h, l);
    int t  = k >> 4, n = k & 15;
    int c  = d >> 5;
    int ls = (d >> 3) & 1;
    int i  = (d & 7) | ((d & 16) >> 1);
    size_t off = ((size_t)(t * 2 + c) * 32 + (ls * 16 + n)) * 16 + i;
    bhi[off] = h;
    blo[off] = l;
}

// Kernel 1: exact f32 ||e_k||^2
__global__ __launch_bounds__(256) void e2_kernel(
    const float* __restrict__ emb, float* __restrict__ e2)
{
    int k = blockIdx.x * blockDim.x + threadIdx.x;
    if (k >= KCODES) return;
    float s = 0.f;
    #pragma unroll
    for (int d = 0; d < DIMD; ++d) {
        float v = emb[d * KCODES + k];
        s = fmaf(v, v, s);
    }
    e2[k] = s;
}

// ---------------------------------------------------------------------------
// Main kernel: each wave handles 32 rows (two 16x16 WMMA C tiles) and scans
// all 4096 codes in 16-code tiles. 12 v_wmma per K-tile (3 split terms x
// 2 D-chunks x 2 row-tiles), B loads shared across both row-tiles.
// ---------------------------------------------------------------------------
__global__ __launch_bounds__(256) void vq_main(
    const float* __restrict__ x, const float* __restrict__ emb,
    const __bf16* __restrict__ bhi, const __bf16* __restrict__ blo,
    const float* __restrict__ e2, float* __restrict__ out,
    float* __restrict__ partials)
{
    const int lane    = threadIdx.x & 31;
    const int wid     = (blockIdx.x * blockDim.x + threadIdx.x) >> 5;  // 0..2047
    const int rowBase = wid * 32;
    const int mrow    = lane & 15;     // C column / A row selector
    const int hsel    = lane >> 4;     // half-wave selector

    // ---- load x rows and split into bf16 hi/lo in the WMMA A layout ----
    // A (16x32 bf16): lane m = lane&15; elem i<8 -> K = hsel*8+i ; i>=8 -> K = 16+hsel*8+(i-8)
    v16bf ahi[2][2], alo[2][2];
    #pragma unroll
    for (int mt = 0; mt < 2; ++mt) {
        const float* xr = x + (size_t)(rowBase + mt * 16 + mrow) * DIMD;
        #pragma unroll
        for (int c = 0; c < 2; ++c) {
            const int base0 = c * 32 + hsel * 8;
            #pragma unroll
            for (int i = 0; i < 8; ++i) {
                __bf16 h, l;
                splitf(xr[base0 + i], h, l);
                ahi[mt][c][i] = h;  alo[mt][c][i] = l;
                splitf(xr[base0 + 16 + i], h, l);
                ahi[mt][c][8 + i] = h;  alo[mt][c][8 + i] = l;
            }
        }
    }

    float minv[2][8];
    int   mini[2][8];
    #pragma unroll
    for (int mt = 0; mt < 2; ++mt)
        #pragma unroll
        for (int j = 0; j < 8; ++j) { minv[mt][j] = 3.4e38f; mini[mt][j] = 0; }

    // ---- scan all codes, 16 per tile ----
    for (int t = 0; t < KCODES / 16; ++t) {
        const size_t boff = ((size_t)(t * 2) * 32 + lane) * 16;
        v16bf bh0 = *(const v16bf*)(bhi + boff);
        v16bf bl0 = *(const v16bf*)(blo + boff);
        v16bf bh1 = *(const v16bf*)(bhi + boff + 512);
        v16bf bl1 = *(const v16bf*)(blo + boff + 512);
        // prefetch next tile's B (emits global_prefetch_b8)
        __builtin_prefetch(bhi + boff + 1024, 0, 1);
        __builtin_prefetch(blo + boff + 1024, 0, 1);

        const int   cand = t * 16 + mrow;
        const float e2v  = e2[cand];

        #pragma unroll
        for (int mt = 0; mt < 2; ++mt) {
            v8f acc = {};
            acc = __builtin_amdgcn_wmma_f32_16x16x32_bf16(false, ahi[mt][0], false, bh0, (short)0, acc, false, false);
            acc = __builtin_amdgcn_wmma_f32_16x16x32_bf16(false, ahi[mt][1], false, bh1, (short)0, acc, false, false);
            acc = __builtin_amdgcn_wmma_f32_16x16x32_bf16(false, ahi[mt][0], false, bl0, (short)0, acc, false, false);
            acc = __builtin_amdgcn_wmma_f32_16x16x32_bf16(false, ahi[mt][1], false, bl1, (short)0, acc, false, false);
            acc = __builtin_amdgcn_wmma_f32_16x16x32_bf16(false, alo[mt][0], false, bh0, (short)0, acc, false, false);
            acc = __builtin_amdgcn_wmma_f32_16x16x32_bf16(false, alo[mt][1], false, bh1, (short)0, acc, false, false);
            // distance (minus row-constant ||f||^2): d = ||e||^2 - 2*sim
            #pragma unroll
            for (int j = 0; j < 8; ++j) {
                float d = fmaf(-2.0f, acc[j], e2v);
                if (d < minv[mt][j]) { minv[mt][j] = d; mini[mt][j] = cand; }
            }
        }
    }

    // ---- per-row argmin across the 16 lanes of each half, then gather ----
    float s1 = 0.f, s2 = 0.f;
    #pragma unroll
    for (int mt = 0; mt < 2; ++mt) {
        #pragma unroll
        for (int j = 0; j < 8; ++j) {
            float v = minv[mt][j];
            int   i = mini[mt][j];
            #pragma unroll
            for (int off = 1; off < 16; off <<= 1) {
                float ov = __shfl_xor(v, off, 32);
                int   oi = __shfl_xor(i, off, 32);
                if (ov < v || (ov == v && oi < i)) { v = ov; i = oi; }
            }
            // lanes 0-15 agree on row mt*16+j ; lanes 16-31 on row mt*16+8+j
            const int row = rowBase + mt * 16 + hsel * 8 + j;
            const int d0  = mrow * 4;   // 16 lanes x 4 dims = 64 dims per row
            float4 q;
            q.x = emb[(size_t)(d0 + 0) * KCODES + i];
            q.y = emb[(size_t)(d0 + 1) * KCODES + i];
            q.z = emb[(size_t)(d0 + 2) * KCODES + i];
            q.w = emb[(size_t)(d0 + 3) * KCODES + i];
            const float4 xv = *(const float4*)(x + (size_t)row * DIMD + d0);
            float dx = q.x - xv.x, dy = q.y - xv.y, dz = q.z - xv.z, dw = q.w - xv.w;
            s1 += (dx + dy) + (dz + dw);
            s2 += fmaf(dx, dx, fmaf(dy, dy, fmaf(dz, dz, dw * dw)));
            // estimator = x + sg(q - x) == q in forward value
            *(float4*)(out + (size_t)row * DIMD + d0) = q;
        }
    }

    // wave-wide sum of loss partials; one slot per wave -> deterministic
    #pragma unroll
    for (int off = 1; off < 32; off <<= 1) {
        s1 += __shfl_xor(s1, off, 32);
        s2 += __shfl_xor(s2, off, 32);
    }
    if (lane == 0) {
        partials[wid]          = s1;
        partials[NWAVES + wid] = s2;
    }
}

// ---------------------------------------------------------------------------
// Final fixed-order reduction: loss = 0.25*mean(q-x)^2 + mean((q-x)^2)
// ---------------------------------------------------------------------------
__global__ __launch_bounds__(1024) void reduce_loss(
    const float* __restrict__ p, float* __restrict__ loss_out)
{
    __shared__ float a[1024];
    __shared__ float b[1024];
    int t = threadIdx.x;
    a[t] = p[t] + p[t + 1024];
    b[t] = p[NWAVES + t] + p[NWAVES + t + 1024];
    __syncthreads();
    for (int s = 512; s > 0; s >>= 1) {
        if (t < s) { a[t] += a[t + s]; b[t] += b[t + s]; }
        __syncthreads();
    }
    if (t == 0) {
        const float invM = 1.0f / (float)MTOTAL;
        float m1 = a[0] * invM;
        loss_out[0] = 0.25f * m1 * m1 + b[0] * invM;   // BETA = 0.25
    }
}

// ---------------------------------------------------------------------------
extern "C" void kernel_launch(void* const* d_in, const int* in_sizes, int n_in,
                              void* d_out, int out_size, void* d_ws, size_t ws_size,
                              hipStream_t stream) {
    const float* x   = (const float*)d_in[0];   // (64,32,32,64) f32
    const float* emb = (const float*)d_in[1];   // (64,4096) f32
    float* out = (float*)d_out;                 // 4194304 estimator + 1 loss

    // workspace layout (bytes):
    //   [0, 512K)        packed B hi  (bf16)
    //   [512K, 1M)       packed B lo  (bf16)
    //   [1M, 1M+16K)     ||e_k||^2    (f32 x 4096)
    //   [1M+16K, +16K)   loss partials (f32 x 2*2048)
    char* ws = (char*)d_ws;
    __bf16* bhi = (__bf16*)(ws);
    __bf16* blo = (__bf16*)(ws + (512u << 10));
    float*  e2  = (float*)(ws + (1024u << 10));
    float*  prt = (float*)(ws + (1024u << 10) + (16u << 10));

    pack_embedding<<<dim3((DIMD * KCODES) / 256), dim3(256), 0, stream>>>(emb, bhi, blo);
    e2_kernel<<<dim3(KCODES / 256), dim3(256), 0, stream>>>(emb, e2);
    // 2048 waves * 32 rows = 65536 rows; 8 waves per 256-thread block
    vq_main<<<dim3(NWAVES / 8), dim3(256), 0, stream>>>(x, emb, bhi, blo, e2, out, prt);
    reduce_loss<<<dim3(1), dim3(1024), 0, stream>>>(prt, out + MTOTAL);
}
